// Attention_23648089932275
// MI455X (gfx1250) — matrix operand order repaired and checked by reading, not compiled, not run on hardware
//
#include <hip/hip_runtime.h>
#include <stdint.h>

#define N_NODES 50000
#define DIM     512
#define KMAX    32
#define BLOCK   256
#define GRID    1024

typedef __attribute__((ext_vector_type(16))) _Float16 v16h;
typedef __attribute__((ext_vector_type(8)))  float    v8f;
typedef __attribute__((ext_vector_type(4)))  int      v4i;

typedef __attribute__((address_space(1))) v4i* gv4i_p;  // global <4 x i32>*
typedef __attribute__((address_space(3))) v4i* lv4i_p;  // LDS    <4 x i32>*

__global__ void init_ws_kernel(int* ws, int n) {
  int t = blockIdx.x * blockDim.x + threadIdx.x;
  if (t < n) ws[t] = 0;
}

__global__ __launch_bounds__(BLOCK) void gat_wavefront_kernel(
    const float* __restrict__ feats,
    const float* __restrict__ wq_w, const float* __restrict__ wq_b,
    const float* __restrict__ wk_w, const float* __restrict__ wk_b,
    const int*   __restrict__ neighbors, const int* __restrict__ deg,
    float* __restrict__ out, int* counter, int* flags)
{
  __shared__ __align__(16) float s_nb[KMAX][DIM];   // 64KB neighbor tile
  __shared__ __align__(16) float s_center[DIM];     // 2KB center row cache
  __shared__ const float* s_rowptr[KMAX];
  __shared__ float s_ej[KMAX];
  __shared__ float s_a[KMAX];
  __shared__ float s_red[BLOCK / 32];
  __shared__ float s_ei;
  __shared__ int   s_node;

  const int tid  = threadIdx.x;
  const int lane = tid & 31;
  const int wave = tid >> 5;
  const float wqb = wq_b[0];
  const float wkb = wk_b[0];

  for (;;) {
    if (tid == 0) s_node = atomicAdd(counter, 1);
    __syncthreads();
    const int i = s_node;
    if (i >= N_NODES) return;
    const int d_i = deg[i];

    // ---- dependency wait + per-row source pointers (wave 0) ----
    if (tid < KMAX) {
      int nbr = neighbors[(size_t)i * KMAX + tid];
      if (tid < d_i && nbr < i) {
        while (__hip_atomic_load(&flags[nbr], __ATOMIC_ACQUIRE,
                                 __HIP_MEMORY_SCOPE_AGENT) == 0)
          __builtin_amdgcn_s_sleep(1);
      }
      // rows < i were updated this launch (live in `out`), rows >= i are original
      s_rowptr[tid] = ((nbr < i) ? out : feats) + (size_t)nbr * DIM;
    }
    __syncthreads();

    // ---- async gather: 32 rows x 2KB -> LDS (16B per lane per op) ----
#pragma unroll
    for (int t = 0; t < 16; ++t) {
      int idx = t * (BLOCK * 4) + tid * 4;        // float index into 32x512
      int k   = idx >> 9;
      int col = idx & (DIM - 1);
      const float* gp = s_rowptr[k] + col;
      float*       lp = &s_nb[k][col];
#if defined(__gfx1250__) && __has_builtin(__builtin_amdgcn_global_load_async_to_lds_b128)
      __builtin_amdgcn_global_load_async_to_lds_b128(
          (gv4i_p)(uintptr_t)gp,
          (lv4i_p)(uint32_t)(uintptr_t)lp, 0, 0);
#else
      *(float4*)lp = *(const float4*)gp;
#endif
    }

    // ---- overlap with async copy: ei = dot(center, wq) + b,
    //      and stash the center row in LDS for the final add ----
    {
      float c0 = feats[(size_t)i * DIM + tid];
      float c1 = feats[(size_t)i * DIM + 256 + tid];
      s_center[tid]       = c0;
      s_center[256 + tid] = c1;
      float p = c0 * wq_w[tid] + c1 * wq_w[256 + tid];
#pragma unroll
      for (int o = 16; o; o >>= 1) p += __shfl_down(p, o, 32);
      if (lane == 0) s_red[wave] = p;
    }

#if defined(__gfx1250__) && __has_builtin(__builtin_amdgcn_s_wait_asynccnt)
    __builtin_amdgcn_s_wait_asynccnt(0);
#elif defined(__gfx1250__)
    asm volatile("s_wait_asynccnt 0" ::: "memory");
#endif
    __syncthreads();

    if (tid == 0) {
      float s = 0.f;
#pragma unroll
      for (int w = 0; w < BLOCK / 32; ++w) s += s_red[w];
      s_ei = s + wqb;
    }

    // ---- ej[k] = dot(nb[k], wk) : 4 rows per wave, float4 DS loads ----
#pragma unroll
    for (int r = 0; r < 4; ++r) {
      int k = wave * 4 + r;
      float p = 0.f;
#pragma unroll
      for (int j = 0; j < 4; ++j) {
        int c = j * 128 + lane * 4;
        float4 nb4 = *(const float4*)&s_nb[k][c];
        float4 wk4 = *(const float4*)&wk_w[c];
        p += nb4.x * wk4.x + nb4.y * wk4.y + nb4.z * wk4.z + nb4.w * wk4.w;
      }
#pragma unroll
      for (int o = 16; o; o >>= 1) p += __shfl_down(p, o, 32);
      if (lane == 0) s_ej[k] = p;
    }
    __syncthreads();

    // ---- masked softmax over K=32 (wave 0) ----
    if (tid < KMAX) {
      float aval = 0.f;
      if (d_i > 0) {
        bool  valid = tid < d_i;
        float e = s_ei * (s_ej[tid] + wkb);
        float x = valid ? e : -3.0e38f;
        float m = x;
#pragma unroll
        for (int o = 16; o; o >>= 1) m = fmaxf(m, __shfl_xor(m, o, 32));
        float ex = valid ? __expf(x - m) : 0.f;
        float s = ex;
#pragma unroll
        for (int o = 16; o; o >>= 1) s += __shfl_xor(s, o, 32);
        aval = ex / s;
      }
      s_a[tid] = aval;
    }
    __syncthreads();

    // ---- upd = a (1x32) @ nb (32x512) via v_wmma_f32_16x16x32_f16 ----
    // A: 16x32, all 16 rows replicated with a[]; lanes0-15 hold K0-7,16-23,
    // lanes16-31 hold K8-15,24-31 (ISA 16-bit A layout).
    {
      const int abase = (lane < 16) ? 0 : 8;
      v16h af;
#pragma unroll
      for (int h = 0; h < 8; ++h) af[h]     = (_Float16)s_a[abase + h];
#pragma unroll
      for (int h = 0; h < 8; ++h) af[8 + h] = (_Float16)s_a[abase + 16 + h];

      const int n  = lane & 15;
      const int kb = (lane < 16) ? 0 : 8;
#pragma unroll
      for (int t = 0; t < 4; ++t) {
        const int d0 = (wave * 4 + t) * 16;
        v16h bf;
#pragma unroll
        for (int h = 0; h < 16; ++h)
          bf[h] = (_Float16)s_nb[kb + h + (h & 8)][d0 + n];
        v8f acc = {};
        acc = __builtin_amdgcn_wmma_f32_16x16x32_f16(
            false, af, false, bf, (short)0, acc, false, false);
        // every M row of D equals upd[d0+n]; lanes 0-15, element 0 is (M=0,N=lane)
        if (lane < 16) {
          out[(size_t)i * DIM + d0 + lane] = s_center[d0 + lane] + acc[0];
        }
      }
    }

    // ---- publish row i ----
    __threadfence();
    __syncthreads();
    if (tid == 0)
      __hip_atomic_store(&flags[i], 1, __ATOMIC_RELEASE, __HIP_MEMORY_SCOPE_AGENT);
  }
}

extern "C" void kernel_launch(void* const* d_in, const int* in_sizes, int n_in,
                              void* d_out, int out_size, void* d_ws, size_t ws_size,
                              hipStream_t stream) {
  (void)in_sizes; (void)n_in; (void)out_size; (void)ws_size;
  const float* feats     = (const float*)d_in[0];
  const float* wq_w      = (const float*)d_in[1];
  const float* wq_b      = (const float*)d_in[2];
  const float* wk_w      = (const float*)d_in[3];
  const float* wk_b      = (const float*)d_in[4];
  const int*   neighbors = (const int*)d_in[5];
  const int*   deg       = (const int*)d_in[6];
  float* out = (float*)d_out;

  int* ws      = (int*)d_ws;
  int* counter = ws;        // ws[0]: work counter
  int* flags   = ws + 64;   // per-node done flags, cacheline away from counter

  const int ninit = N_NODES + 64;
  init_ws_kernel<<<(ninit + 255) / 256, 256, 0, stream>>>(ws, ninit);
  gat_wavefront_kernel<<<GRID, BLOCK, 0, stream>>>(
      feats, wq_w, wq_b, wk_w, wk_b, neighbors, deg, out, counter, flags);
}
